// GIN_classifier_90443421319566
// MI455X (gfx1250) — matrix operand order repaired and checked by reading, not compiled, not run on hardware
//
#include <hip/hip_runtime.h>

#define N_NODES 50000
#define N_EDGES 600000
#define DIN     128
#define HID     128
#define BN_EPS  1e-5f

typedef __attribute__((ext_vector_type(16))) __bf16 v16bf;
typedef __attribute__((ext_vector_type(8)))  float  v8f;

__device__ __forceinline__ __bf16 tobf(float f) { return (__bf16)f; }

// ---------------------------------------------------------------------------
// 1) Prep: W1t/W2t = bf16 transpose (N-major) of W1/W2; zero BN accumulators.
// ---------------------------------------------------------------------------
__global__ void gin_prep_kernel(const float* __restrict__ W1,
                                const float* __restrict__ W2,
                                __bf16* __restrict__ W1t,
                                __bf16* __restrict__ W2t,
                                float* __restrict__ colsum,
                                float* __restrict__ colsq) {
  int t = blockIdx.x * blockDim.x + threadIdx.x;     // 0 .. 16383
  if (t < HID * HID) {
    int n = t / HID, k = t % HID;                    // Wt[n][k] = W[k][n]
    W1t[t] = tobf(W1[k * HID + n]);
    W2t[t] = tobf(W2[k * HID + n]);
  }
  if (t < HID) { colsum[t] = 0.f; colsq[t] = 0.f; }
}

// ---------------------------------------------------------------------------
// 2) agg = x  (GIN self term, eps=0)
// ---------------------------------------------------------------------------
__global__ void gin_copy_kernel(const float* __restrict__ x,
                                float* __restrict__ agg) {
  size_t i = (size_t)blockIdx.x * blockDim.x + threadIdx.x;
  if (i < (size_t)N_NODES * DIN / 4)
    ((float4*)agg)[i] = ((const float4*)x)[i];
}

// ---------------------------------------------------------------------------
// 3) Edge scatter: one wave32 per edge, each lane handles 4 contiguous floats.
//    x fits in 192MB L2, so gathers + f32 atomic RMW stay on-chip.
// ---------------------------------------------------------------------------
__global__ void gin_scatter_kernel(const float* __restrict__ x,
                                   const int* __restrict__ eidx,
                                   float* __restrict__ agg) {
  int wave = (int)((blockIdx.x * blockDim.x + threadIdx.x) >> 5);
  int lane = threadIdx.x & 31;
  if (wave >= N_EDGES) return;
  int src = eidx[wave];
  int dst = eidx[N_EDGES + wave];
  float4 v = ((const float4*)(x + (size_t)src * DIN))[lane];
  float* d = agg + (size_t)dst * DIN + lane * 4;
  atomicAdd(d + 0, v.x);
  atomicAdd(d + 1, v.y);
  atomicAdd(d + 2, v.z);
  atomicAdd(d + 3, v.w);
}

// ---------------------------------------------------------------------------
// 4/6) WMMA GEMM: C[M,128] = op(A)[M,128] @ Wt^T + bias
//   MODE 0: op(A)=A,               out = relu(.), accumulate BN sum/sumsq
//   MODE 1: op(A)=A*scale+shift,   out = plain
// Block: 256 threads = 8 waves; 128 rows per block; 8 N-tiles of 16; K = 4x32.
// ---------------------------------------------------------------------------
template <int MODE>
__global__ void gin_gemm_kernel(const float* __restrict__ A,
                                const __bf16* __restrict__ Wt,
                                const float* __restrict__ bias,
                                const float* __restrict__ scale,
                                const float* __restrict__ shift,
                                float* __restrict__ Out,
                                float* __restrict__ colsum,
                                float* __restrict__ colsq) {
  __shared__ float lsum[HID];
  __shared__ float lsq[HID];
  if (MODE == 0) {
    if (threadIdx.x < HID) { lsum[threadIdx.x] = 0.f; lsq[threadIdx.x] = 0.f; }
    __syncthreads();
  }

  const int lane = threadIdx.x & 31;
  const int wv   = threadIdx.x >> 5;    // wave 0..7 -> 16 rows each
  const int half = lane >> 4;           // 0 | 1
  const int l16  = lane & 15;

  // A-operand row held by this lane (ISA 16-bit A 16x32 layout)
  int rowA = blockIdx.x * 128 + wv * 16 + l16;
  int rowA_c = rowA < N_NODES ? rowA : (N_NODES - 1);   // clamp; guarded later
  const float* arow = A + (size_t)rowA_c * HID;

  v8f acc[8];
#pragma unroll
  for (int t = 0; t < 8; ++t) acc[t] = (v8f){0.f,0.f,0.f,0.f,0.f,0.f,0.f,0.f};

#pragma unroll
  for (int kk = 0; kk < 4; ++kk) {
    const int k0 = kk * 32;
    const int b0 = k0 + half * 16;       // K = b0..b0+7
    const int b1 = k0 + half * 16 + 8;   // K = b1..b1+7
    v16bf afrag;
#pragma unroll
    for (int j = 0; j < 8; ++j) {
      float f0 = arow[b0 + j];
      float f1 = arow[b1 + j];
      if (MODE == 1) {
        f0 = f0 * scale[b0 + j] + shift[b0 + j];
        f1 = f1 * scale[b1 + j] + shift[b1 + j];
      }
      afrag[j]     = tobf(f0);
      afrag[8 + j] = tobf(f1);
    }
#pragma unroll
    for (int t = 0; t < 8; ++t) {
      const int n = t * 16 + l16;
      // B fragment: 16 contiguous K values of column n (transposed weights)
      v16bf bfrag = *(const v16bf*)(Wt + n * HID + k0 + half * 16);
      acc[t] = __builtin_amdgcn_wmma_f32_16x16x32_bf16(
          false, afrag, false, bfrag, (short)0, acc[t], false, false);
    }
  }

  // C/D layout: col n = t*16 + l16; rows = base + half*8 + i
  const int rBase = blockIdx.x * 128 + wv * 16 + half * 8;
#pragma unroll
  for (int t = 0; t < 8; ++t) {
    const int n = t * 16 + l16;
    const float b = bias[n];
    float psum = 0.f, psq = 0.f;
#pragma unroll
    for (int i = 0; i < 8; ++i) {
      const int r = rBase + i;
      if (r < N_NODES) {
        float v = acc[t][i] + b;
        if (MODE == 0) { v = fmaxf(v, 0.f); psum += v; psq += v * v; }
        Out[(size_t)r * HID + n] = v;
      }
    }
    if (MODE == 0) {
      atomicAdd(&lsum[n], psum);
      atomicAdd(&lsq[n], psq);
    }
  }
  if (MODE == 0) {
    __syncthreads();
    if (threadIdx.x < HID) {
      atomicAdd(&colsum[threadIdx.x], lsum[threadIdx.x]);
      atomicAdd(&colsq[threadIdx.x],  lsq[threadIdx.x]);
    }
  }
}

// ---------------------------------------------------------------------------
// 5) BN params from batch stats: scale = gamma*rsqrt(var+eps), shift = beta - mean*scale
// ---------------------------------------------------------------------------
__global__ void gin_bnparam_kernel(const float* __restrict__ colsum,
                                   const float* __restrict__ colsq,
                                   const float* __restrict__ gamma,
                                   const float* __restrict__ beta,
                                   float* __restrict__ scale,
                                   float* __restrict__ shift) {
  int c = threadIdx.x;
  if (c < HID) {
    const float invN = 1.0f / (float)N_NODES;
    float mean = colsum[c] * invN;
    float var  = colsq[c] * invN - mean * mean;
    float s    = rsqrtf(var + BN_EPS) * gamma[c];
    scale[c] = s;
    shift[c] = beta[c] - mean * s;
  }
}

// ---------------------------------------------------------------------------
// 7) Head: out[M,2] = H @ Wl + bl.  One wave per row, shuffle reduction.
// ---------------------------------------------------------------------------
__global__ void gin_head_kernel(const float* __restrict__ H,
                                const float* __restrict__ Wl,
                                const float* __restrict__ bl,
                                float* __restrict__ out) {
  int row  = (int)((blockIdx.x * blockDim.x + threadIdx.x) >> 5);
  int lane = threadIdx.x & 31;
  if (row >= N_NODES) return;
  float4 v = ((const float4*)(H + (size_t)row * HID))[lane];
  const float2* W = (const float2*)Wl;           // Wl is [128,2] row-major
  float2 w0 = W[lane * 4 + 0];
  float2 w1 = W[lane * 4 + 1];
  float2 w2 = W[lane * 4 + 2];
  float2 w3 = W[lane * 4 + 3];
  float p0 = v.x * w0.x + v.y * w1.x + v.z * w2.x + v.w * w3.x;
  float p1 = v.x * w0.y + v.y * w1.y + v.z * w2.y + v.w * w3.y;
#pragma unroll
  for (int off = 16; off >= 1; off >>= 1) {
    p0 += __shfl_xor(p0, off, 32);
    p1 += __shfl_xor(p1, off, 32);
  }
  if (lane == 0) {
    out[(size_t)row * 2 + 0] = p0 + bl[0];
    out[(size_t)row * 2 + 1] = p1 + bl[1];
  }
}

// ---------------------------------------------------------------------------
extern "C" void kernel_launch(void* const* d_in, const int* in_sizes, int n_in,
                              void* d_out, int out_size, void* d_ws, size_t ws_size,
                              hipStream_t stream) {
  const float* x     = (const float*)d_in[0];
  const int*   eidx  = (const int*)  d_in[1];
  const float* W1    = (const float*)d_in[2];
  const float* b1    = (const float*)d_in[3];
  const float* gamma = (const float*)d_in[4];
  const float* beta  = (const float*)d_in[5];
  const float* W2    = (const float*)d_in[6];
  const float* b2    = (const float*)d_in[7];
  const float* Wl    = (const float*)d_in[8];
  const float* bl    = (const float*)d_in[9];
  float* out = (float*)d_out;

  // Workspace layout (bytes)
  char* ws = (char*)d_ws;
  const size_t FEAT = (size_t)N_NODES * HID * sizeof(float);   // 25.6 MB
  float*  agg    = (float*)ws;                                  // also reused as h2
  float*  h1     = (float*)(ws + FEAT);
  __bf16* W1t    = (__bf16*)(ws + 2 * FEAT);
  __bf16* W2t    = (__bf16*)(ws + 2 * FEAT + 32768);
  float*  colsum = (float*)(ws + 2 * FEAT + 65536);
  float*  colsq  = colsum + 128;
  float*  scale  = colsum + 256;
  float*  shift  = colsum + 384;

  // 1) weight conversion + stat zeroing
  gin_prep_kernel<<<64, 256, 0, stream>>>(W1, W2, W1t, W2t, colsum, colsq);
  // 2) agg = x
  gin_copy_kernel<<<(N_NODES * DIN / 4 + 255) / 256, 256, 0, stream>>>(x, agg);
  // 3) scatter-add over edges (wave per edge)
  gin_scatter_kernel<<<(N_EDGES * 32 + 255) / 256, 256, 0, stream>>>(x, eidx, agg);
  // 4) h1 = relu(agg @ W1 + b1), BN stats
  gin_gemm_kernel<0><<<(N_NODES + 127) / 128, 256, 0, stream>>>(
      agg, W1t, b1, nullptr, nullptr, h1, colsum, colsq);
  // 5) BN scale/shift
  gin_bnparam_kernel<<<1, 128, 0, stream>>>(colsum, colsq, gamma, beta, scale, shift);
  // 6) h2 = ((h1 - mean)*inv*gamma + beta) @ W2 + b2   (into agg buffer)
  gin_gemm_kernel<1><<<(N_NODES + 127) / 128, 256, 0, stream>>>(
      h1, W2t, b2, scale, shift, agg, nullptr, nullptr);
  // 7) out = h2 @ Wl + bl
  gin_head_kernel<<<(N_NODES * 32 + 255) / 256, 256, 0, stream>>>(agg, Wl, bl, out);
}
